// SelfAttention_32598801777040
// MI455X (gfx1250) — matrix operand order repaired and checked by reading, not compiled, not run on hardware
//
#include <hip/hip_runtime.h>

// ---------------------------------------------------------------------------
// Self-attention (B=4, C=128, N=64*64=4096, d=16), flash-attention style.
//   q = Wq x + bq [B,N,16]   k = Wk x + bk [B,16,N]   v = Wv x + bv [B,C,N]
//   S^T = K Q^T (over d),  P = softmax_j,  out = gamma * (V P) + x
// S is computed TRANSPOSED so the softmax reduction over keys j is in-lane
// (j = vgpr + 8*half) and the per-query running max/sum are per-lane scalars
// (i = lane%16). Each wave owns one 16-query tile and all 128 channels:
// no barriers in the main loop. Matmuls: v_wmma_f32_16x16x32_bf16.
// ---------------------------------------------------------------------------

#define BB 4
#define CC 128
#define NN 4096
#define DD 16

typedef __attribute__((ext_vector_type(16))) __bf16 v16bf;
typedef __attribute__((ext_vector_type(8)))  float  v8f;

union ABf {
    v16bf v;
    unsigned int u[8];
};

__device__ __forceinline__ unsigned short f32_to_bf16(float f) {
    unsigned int u = __float_as_uint(f);
    unsigned int r = 0x7FFFu + ((u >> 16) & 1u);  // round-to-nearest-even
    return (unsigned short)((u + r) >> 16);
}

__device__ __forceinline__ unsigned int pack_bf16(float lo, float hi) {
    return (unsigned int)f32_to_bf16(lo) | ((unsigned int)f32_to_bf16(hi) << 16);
}

// ---------------------------------------------------------------------------
// Projection kernels (1x1 convs): tiny fraction of total FLOPs, VALU is fine.
// ---------------------------------------------------------------------------
__global__ __launch_bounds__(128) void proj_qk_kernel(
    const float* __restrict__ x,  const float* __restrict__ wq,
    const float* __restrict__ bq, const float* __restrict__ wk,
    const float* __restrict__ bk,
    unsigned short* __restrict__ Qg, unsigned short* __restrict__ Kg)
{
    int t = blockIdx.x * blockDim.x + threadIdx.x;   // token id over B*N
    if (t >= BB * NN) return;
    int b = t / NN, n = t % NN;
    const float* xp = x + (size_t)b * CC * NN + n;

    float qa[DD], ka[DD];
    #pragma unroll
    for (int d = 0; d < DD; ++d) { qa[d] = bq[d]; ka[d] = bk[d]; }

    #pragma unroll 4
    for (int c = 0; c < CC; ++c) {
        float xv = xp[(size_t)c * NN];
        #pragma unroll
        for (int d = 0; d < DD; ++d) {
            qa[d] = fmaf(xv, wq[d * CC + c], qa[d]);
            ka[d] = fmaf(xv, wk[d * CC + c], ka[d]);
        }
    }
    #pragma unroll
    for (int d = 0; d < DD; ++d) {
        Qg[(size_t)t * DD + d] = f32_to_bf16(qa[d]);
        Kg[(size_t)t * DD + d] = f32_to_bf16(ka[d]);
    }
}

__global__ __launch_bounds__(256) void proj_v_kernel(
    const float* __restrict__ x, const float* __restrict__ wv,
    const float* __restrict__ bv, unsigned short* __restrict__ Vg)
{
    int e    = blockIdx.x;          // (b, c, n-block of 256)
    int nblk = e & 15;              // NN/256 = 16
    int c    = (e >> 4) & (CC - 1);
    int b    = e >> 11;
    int n    = nblk * 256 + threadIdx.x;

    const float* xp = x + (size_t)b * CC * NN + n;
    float acc = bv[c];
    #pragma unroll 4
    for (int cc = 0; cc < CC; ++cc)
        acc = fmaf(xp[(size_t)cc * NN], wv[c * CC + cc], acc);
    Vg[((size_t)b * CC + c) * NN + n] = f32_to_bf16(acc);
}

// ---------------------------------------------------------------------------
// Fused flash attention: 4 independent wave32 per block, 16 queries per wave,
// each wave accumulates all 128 channels (8 WMMA C/D tiles).
// ---------------------------------------------------------------------------
__global__ __launch_bounds__(128) void attn_kernel(
    const unsigned short* __restrict__ Qg,
    const unsigned short* __restrict__ Kg,
    const unsigned short* __restrict__ Vg,
    const float* __restrict__ x,
    const float* __restrict__ gamma,
    float* __restrict__ out)
{
    // Per-wave P staging: P[i][j] bf16, i-major (row = 32 ushorts = 64 B).
    __shared__ unsigned short Pw[4 * 16 * 32];

    const int tid  = threadIdx.x;
    const int wave = tid >> 5;
    const int lane = tid & 31;
    const int half = lane >> 4;
    const int lr   = lane & 15;

    const int b    = blockIdx.x >> 6;        // NN/64 = 64 query-blocks / batch
    const int qblk = blockIdx.x & 63;
    const int i0   = qblk * 64 + wave * 16;  // this wave's 16 queries

    unsigned short* Pb = Pw + wave * 512 + lr * 32;   // this lane's P row (i=lr)

    // ---- Q B-tile for S^T = K Q^T: B[K=d pad32][N=i] -----------------------
    // B layout: col N = lane%16; VGPR v0..3: K(d) = 8*half + 2v,2v+1 (16B load);
    // v4..7: pad zeros.
    ABf qb;
    {
        const uint4* qr =
            reinterpret_cast<const uint4*>(Qg + ((size_t)(b * NN + i0 + lr)) * DD);
        uint4 a = qr[half];
        qb.u[0] = a.x;  qb.u[1] = a.y;  qb.u[2] = a.z;  qb.u[3] = a.w;
        #pragma unroll
        for (int v = 4; v < 8; ++v) qb.u[v] = 0;
    }

    v8f O[8];
    #pragma unroll
    for (int t = 0; t < 8; ++t) O[t] = (v8f){0.f,0.f,0.f,0.f,0.f,0.f,0.f,0.f};
    float m_run = -__builtin_inff();   // running max for query i = i0+lr
    float l_run = 0.f;                 // running denominator

    for (int jc = 0; jc < NN / 32; ++jc) {
        const int j0 = jc * 32;

        // ---- K A-tiles (16x32, d pad): row M = j_local = lane%16;
        // VGPR v0..3: K(d) = 8*half + 2v,2v+1 (16B load); v4..7: pad zero.
        ABf ka0, ka1;
        {
            const uint4* k0 =
                reinterpret_cast<const uint4*>(Kg + ((size_t)(b * NN + j0 + lr)) * DD);
            const uint4* k1 =
                reinterpret_cast<const uint4*>(Kg + ((size_t)(b * NN + j0 + 16 + lr)) * DD);
            uint4 a = k0[half], c4 = k1[half];
            ka0.u[0] = a.x;  ka0.u[1] = a.y;  ka0.u[2] = a.z;  ka0.u[3] = a.w;
            ka1.u[0] = c4.x; ka1.u[1] = c4.y; ka1.u[2] = c4.z; ka1.u[3] = c4.w;
            #pragma unroll
            for (int v = 4; v < 8; ++v) { ka0.u[v] = 0; ka1.u[v] = 0; }
        }

        v8f zc = {0.f,0.f,0.f,0.f,0.f,0.f,0.f,0.f};
        // S^T tiles: row = j_local = v + 8*half, col = i = lane%16
        v8f S0 = __builtin_amdgcn_wmma_f32_16x16x32_bf16(
                     false, ka0.v, false, qb.v, (short)0, zc, false, false);
        v8f S1 = __builtin_amdgcn_wmma_f32_16x16x32_bf16(
                     false, ka1.v, false, qb.v, (short)0, zc, false, false);

        // ---- online softmax over j: in-lane over 8 VGPRs, 1 shuffle -------
        float cm = fmaxf(S0[0], S1[0]);
        #pragma unroll
        for (int v = 1; v < 8; ++v) cm = fmaxf(cm, fmaxf(S0[v], S1[v]));
        cm = fmaxf(cm, __shfl_xor(cm, 16, 32));          // merge lane halves
        float mn = fmaxf(m_run, cm);
        float sc = __expf(m_run - mn);                   // 0 on first chunk
        float e0[8], e1[8], rs = 0.f;
        #pragma unroll
        for (int v = 0; v < 8; ++v) {
            e0[v] = __expf(S0[v] - mn);
            e1[v] = __expf(S1[v] - mn);
            rs += e0[v] + e1[v];
        }
        rs += __shfl_xor(rs, 16, 32);
        l_run = l_run * sc + rs;
        m_run = mn;

        // ---- stage P in LDS (bf16, i-major): lane holds j = 8*half + v
        // (tile0) and j = 16 + 8*half + v (tile1) -> two 16B stores.
        {
            uint4 w0, w1;
            w0.x = pack_bf16(e0[0], e0[1]); w0.y = pack_bf16(e0[2], e0[3]);
            w0.z = pack_bf16(e0[4], e0[5]); w0.w = pack_bf16(e0[6], e0[7]);
            w1.x = pack_bf16(e1[0], e1[1]); w1.y = pack_bf16(e1[2], e1[3]);
            w1.z = pack_bf16(e1[4], e1[5]); w1.w = pack_bf16(e1[6], e1[7]);
            *reinterpret_cast<uint4*>(Pb + 8 * half)      = w0;
            *reinterpret_cast<uint4*>(Pb + 16 + 8 * half) = w1;
        }
        // Wave-internal cross-lane RAW through LDS: DS is in-order per wave;
        // wait for the stores, and fence the compiler.
        asm volatile("s_wait_dscnt 0x0" ::: "memory");

        // ---- P B-tile (32x16): col N = i = lane%16; VGPR v<4: K(j)=8h+2v,2v+1; v>=4: K(j)=16+8h+2(v-4),+1
        ABf pb;
        #pragma unroll
        for (int v = 0; v < 8; ++v)
            pb.u[v] = *reinterpret_cast<const unsigned int*>(Pb + 8 * half + 2 * v + (v >= 4 ? 8 : 0));

        // ---- rescale running output, then O[t] += V_tile x P ---------------
        #pragma unroll
        for (int t = 0; t < 8; ++t) {
            #pragma unroll
            for (int v = 0; v < 8; ++v) O[t][v] *= sc;
        }
        #pragma unroll
        for (int t = 0; t < 8; ++t) {
            // V A-tile: row M = c = 16t + lane%16; VGPR0..3: j=j0+8h+0..7,
            // VGPR4..7: j=j0+16+8h+0..7 (two 16B loads).
            const unsigned short* vr =
                Vg + (((size_t)b * CC + t * 16 + lr) * NN + j0 + half * 8);
            uint4 A0 = *reinterpret_cast<const uint4*>(vr);
            uint4 A1 = *reinterpret_cast<const uint4*>(vr + 16);
            ABf va;
            va.u[0] = A0.x; va.u[1] = A0.y; va.u[2] = A0.z; va.u[3] = A0.w;
            va.u[4] = A1.x; va.u[5] = A1.y; va.u[6] = A1.z; va.u[7] = A1.w;
            __builtin_prefetch(vr + 32, 0, 1);   // next j-chunk of this row
            O[t] = __builtin_amdgcn_wmma_f32_16x16x32_bf16(
                       false, va.v, false, pb.v, (short)0, O[t], false, false);
        }
    }

    // ---- epilogue: out = gamma * O/l + x -----------------------------------
    float linv = 1.0f / l_run;           // per-lane: query i = i0 + lr
    float g    = gamma[0];
    #pragma unroll
    for (int t = 0; t < 8; ++t) {
        #pragma unroll
        for (int v = 0; v < 8; ++v) {
            int c = t * 16 + v + 8 * half;   // D layout: row M = v + 8*half
            size_t idx = ((size_t)b * CC + c) * NN + i0 + lr;
            out[idx] = fmaf(g, O[t][v] * linv, x[idx]);
        }
    }
}

// ---------------------------------------------------------------------------
extern "C" void kernel_launch(void* const* d_in, const int* in_sizes, int n_in,
                              void* d_out, int out_size, void* d_ws, size_t ws_size,
                              hipStream_t stream)
{
    const float* x     = (const float*)d_in[0];
    const float* wq    = (const float*)d_in[1];
    const float* bq    = (const float*)d_in[2];
    const float* wk    = (const float*)d_in[3];
    const float* bk    = (const float*)d_in[4];
    const float* wv    = (const float*)d_in[5];
    const float* bv    = (const float*)d_in[6];
    const float* gamma = (const float*)d_in[7];
    float* out = (float*)d_out;

    // workspace: Q [B*N*16] bf16, K [B*N*16] bf16, V [B*C*N] bf16 (~5 MB)
    unsigned short* Qg = (unsigned short*)d_ws;
    unsigned short* Kg = Qg + (size_t)BB * NN * DD;
    unsigned short* Vg = Kg + (size_t)BB * NN * DD;

    proj_qk_kernel<<<(BB * NN + 127) / 128, 128, 0, stream>>>(x, wq, bq, wk, bk, Qg, Kg);
    proj_v_kernel<<<BB * CC * (NN / 256), 256, 0, stream>>>(x, wv, bv, Vg);
    attn_kernel<<<BB * (NN / 64), 128, 0, stream>>>(Qg, Kg, Vg, x, gamma, out);
}
